// DynaResidualBlockC_65343632442107
// MI455X (gfx1250) — matrix operand order repaired and checked by reading, not compiled, not run on hardware
//
#include <hip/hip_runtime.h>

typedef __attribute__((ext_vector_type(16))) _Float16 v16h;
typedef __attribute__((ext_vector_type(8)))  _Float16 v8h;
typedef __attribute__((ext_vector_type(8)))  float    v8f;
typedef __attribute__((ext_vector_type(4)))  float    v4f;

#define NB     16          // batches
#define NLAT   512
#define NFIN   64
#define NFOUT  64
#define HWPIX  36864       // 192*192
#define WPB    24576       // f16 weight elements per batch in ws
// fragment region bases (in halfs, per batch); each fragment = 512 halfs
#define FRAG_IN     0      // 4 mtiles x 2 ktiles
#define FRAG_MID    4096   // 4 x 4
#define FRAG_OUT    12288  // 4 x 4
#define FRAG_SHORT  20480  // 4 x 2
#define BIAS_OFF_BYTES 786432   // after 16*24576 halfs

// ---------------------------------------------------------------------------
// Kernel 1: ks = lat @ Wk.T + bk  -> scaled f16 weights scattered into WMMA
// A-fragment layout (16-bit A 16x32, this code's contiguous k = 16*(lane/16) + t order), plus f32 biases.
// One thread per output row j, all 16 batches per thread.
// ---------------------------------------------------------------------------
__global__ __launch_bounds__(256)
void dyn_weights_kernel(const float* __restrict__ lat,   // [NB][NLAT]
                        const float* __restrict__ Wk,    // [KTOT][NLAT]
                        const float* __restrict__ bk,    // [KTOT]
                        _Float16*    __restrict__ wfrag, // [NB][WPB]
                        float*       __restrict__ bias)  // [NB][256]
{
    __shared__ float latT[NLAT * NB];   // [l][bb] -> 16 f32 contiguous per l
    const int tid = threadIdx.x;
    for (int idx = tid; idx < NB * NLAT; idx += 256) {
        int bb = idx >> 9;
        int l  = idx & (NLAT - 1);
        latT[l * NB + bb] = lat[idx];
    }
    __syncthreads();

    const int j = blockIdx.x * 256 + tid;          // grid 97*256 == 24832

    float acc[NB];
#pragma unroll
    for (int bb = 0; bb < NB; ++bb) acc[bb] = 0.f;

    const float* wrow = Wk + (size_t)j * NLAT;
    for (int l = 0; l < NLAT; l += 4) {
        v4f w4 = *(const v4f*)(wrow + l);
#pragma unroll
        for (int q = 0; q < 4; ++q) {
            const float wv = w4[q];
            const float* lt = &latT[(l + q) * NB];
            v4f a0 = *(const v4f*)(lt + 0);
            v4f a1 = *(const v4f*)(lt + 4);
            v4f a2 = *(const v4f*)(lt + 8);
            v4f a3 = *(const v4f*)(lt + 12);
#pragma unroll
            for (int r = 0; r < 4; ++r) {
                acc[r +  0] = fmaf(wv, a0[r], acc[r +  0]);
                acc[r +  4] = fmaf(wv, a1[r], acc[r +  4]);
                acc[r +  8] = fmaf(wv, a2[r], acc[r +  8]);
                acc[r + 12] = fmaf(wv, a3[r], acc[r + 12]);
            }
        }
    }
    const float bkj = bk[j];

    if (j >= 24576) {                               // bias rows -> f32
        const int bl = j - 24576;                   // 0..255
#pragma unroll
        for (int bb = 0; bb < NB; ++bb)
            bias[bb * 256 + bl] = acc[bb] + bkj;
        return;
    }

    int jl, base, nkt, kshift;
    float scale;
    if (j < 4096)        { jl = j;          base = FRAG_IN;    nkt = 2; kshift = 6; scale = 0.0883883476483184f; }
    else if (j < 12288)  { jl = j - 4096;   base = FRAG_MID;   nkt = 4; kshift = 7; scale = 0.0883883476483184f; }
    else if (j < 20480)  { jl = j - 12288;  base = FRAG_OUT;   nkt = 4; kshift = 7; scale = 0.125f; }
    else                 { jl = j - 20480;  base = FRAG_SHORT; nkt = 2; kshift = 6; scale = 0.125f; }

    const int m  = jl >> kshift;
    const int k  = jl & ((1 << kshift) - 1);
    const int mt = m >> 4;
    const int kt = k >> 5;
    const int kk = k & 31;
    const int g    = kk >> 4;                       // half-wave group
    const int lane = (m & 15) | (g << 4);
    const int t    = kk & 15;                       // half index within lane's 32B
    const int addr = base + (mt * nkt + kt) * 512 + lane * 16 + t;

#pragma unroll
    for (int bb = 0; bb < NB; ++bb)
        wfrag[bb * WPB + addr] = (_Float16)((acc[bb] + bkj) * scale);
}

// ---------------------------------------------------------------------------
// Kernel 2: per-pixel dynamic MLP with cos/sin expansion, all matmuls as
// v_wmma_f32_16x16x32_f16. One wave owns TWO 16-pixel subtiles (32 pixels),
// reusing each A fragment for 2 WMMAs. C->B relayout via per-wave LDS.
// Bias table staged into LDS via async global->LDS DMA (inline asm).
// ---------------------------------------------------------------------------
__global__ __launch_bounds__(128)
void dyn_apply_kernel(const float*    __restrict__ x,     // [NB][64][HWPIX]
                      const _Float16* __restrict__ wfrag, // [NB][WPB]
                      const float*    __restrict__ bias,  // [NB][256]
                      float*          __restrict__ out)   // [NB][64][HWPIX]
{
    __shared__ _Float16 zbuf[4][32][144];   // per-wave [pixel][k] f16, 288B row stride
    __shared__ float    bias_lds[256];

    const int b    = blockIdx.x;
    const int tid  = threadIdx.x;
    const int wave = tid >> 5;
    const int lane = tid & 31;
    const int n    = lane & 15;             // column (pixel) within a subtile
    const int g    = lane >> 4;             // half-wave group
    const int pixbase = blockIdx.y * 128 + wave * 32;

    const _Float16* wb = wfrag + (size_t)b * WPB;
    const float*    xb = x   + (size_t)b * NFIN  * HWPIX;
    float*          ob = out + (size_t)b * NFOUT * HWPIX;
    const float*    bias_b = bias + b * 256;

    // ---- stage per-batch bias table (1KB) into LDS via async DMA ----
    // 64 lanes x 16B = 1KB; VDST = LDS byte offset (low 32b of generic ptr),
    // VADDR = 64-bit global address, no SADDR.
    if (tid < 64) {
        unsigned lds_off = (unsigned)(unsigned long long)(&bias_lds[0]) + tid * 16u;
        unsigned long long gaddr =
            (unsigned long long)((const char*)bias_b + tid * 16);
        asm volatile("global_load_async_to_lds_b128 %0, %1, off"
                     :: "v"(lds_off), "v"(gaddr) : "memory");
    }
    asm volatile("s_wait_asynccnt 0x0" ::: "memory");
    __syncthreads();

    // pull this batch's weight block toward the WGP caches
    __builtin_prefetch(wb + lane * 64, 0, 3);

    auto ldA = [&](int fragbase, int idx) -> v16h {
        return *(const v16h*)(wb + fragbase + idx * 512 + lane * 16);
    };

    const v8f vzero = {0.f, 0.f, 0.f, 0.f, 0.f, 0.f, 0.f, 0.f};

    // ---- load X tiles in B-layout (K = channel): 2 subtiles x 2 K-chunks ----
    v16h bx[2][2];
#pragma unroll
    for (int nt = 0; nt < 2; ++nt) {
#pragma unroll
        for (int c = 0; c < 2; ++c) {
#pragma unroll
            for (int t = 0; t < 16; ++t) {
                const int ch = 32 * c + 16 * g + t;
                bx[nt][c][t] = (_Float16)xb[(size_t)ch * HWPIX + pixbase + nt * 16 + n];
            }
        }
    }

    // ---- GEMM: k_short @ x  and  k_in @ x ----
    v8f acc_s[2][4] = {{vzero, vzero, vzero, vzero}, {vzero, vzero, vzero, vzero}};
    v8f acc_y[2][4] = {{vzero, vzero, vzero, vzero}, {vzero, vzero, vzero, vzero}};
#pragma unroll
    for (int c = 0; c < 2; ++c) {
#pragma unroll
        for (int a = 0; a < 4; ++a) {
            v16h as = ldA(FRAG_SHORT, a * 2 + c);
#pragma unroll
            for (int nt = 0; nt < 2; ++nt)
                acc_s[nt][a] = __builtin_amdgcn_wmma_f32_16x16x32_f16(
                    false, as, false, bx[nt][c], (short)0, acc_s[nt][a], false, false);
            v16h ai = ldA(FRAG_IN, a * 2 + c);
#pragma unroll
            for (int nt = 0; nt < 2; ++nt)
                acc_y[nt][a] = __builtin_amdgcn_wmma_f32_16x16x32_f16(
                    false, ai, false, bx[nt][c], (short)0, acc_y[nt][a], false, false);
        }
    }

    // ---- stage 1 wave: z = [cos(y+b_in), sin(y+b_in)] -> LDS in B-layout ----
#pragma unroll
    for (int nt = 0; nt < 2; ++nt) {
        _Float16* zr = &zbuf[wave][nt * 16 + n][0];
#pragma unroll
        for (int a = 0; a < 4; ++a) {
            const int m0 = 16 * a + 8 * g;          // first row held by this lane
            v4f b0 = *(const v4f*)(bias_lds + m0);
            v4f b1 = *(const v4f*)(bias_lds + m0 + 4);
            v8h cz, sz;
#pragma unroll
            for (int r = 0; r < 8; ++r) {
                const float bv = (r < 4) ? b0[r] : b1[r - 4];
                const float yv = acc_y[nt][a][r] + bv;
                cz[r] = (_Float16)__cosf(yv);
                sz[r] = (_Float16)__sinf(yv);
            }
            *(v8h*)(zr + m0)      = cz;             // k = m
            *(v8h*)(zr + 64 + m0) = sz;             // k = 64 + m
        }
    }

    // ---- GEMM: k_mid @ z  (K = 128 -> 4 chunks) ----
    v16h bz[2][4];
#pragma unroll
    for (int nt = 0; nt < 2; ++nt)
#pragma unroll
        for (int c = 0; c < 4; ++c)
            bz[nt][c] = *(const v16h*)(&zbuf[wave][nt * 16 + n][32 * c + 16 * g]);

#pragma unroll
    for (int nt = 0; nt < 2; ++nt)
#pragma unroll
        for (int a = 0; a < 4; ++a) acc_y[nt][a] = vzero;
#pragma unroll
    for (int c = 0; c < 4; ++c) {
#pragma unroll
        for (int a = 0; a < 4; ++a) {
            v16h am = ldA(FRAG_MID, a * 4 + c);
#pragma unroll
            for (int nt = 0; nt < 2; ++nt)
                acc_y[nt][a] = __builtin_amdgcn_wmma_f32_16x16x32_f16(
                    false, am, false, bz[nt][c], (short)0, acc_y[nt][a], false, false);
        }
    }

    // ---- stage 2 wave: z2 = [cos(y+b_mid), sin(y+b_mid)] -> LDS ----
#pragma unroll
    for (int nt = 0; nt < 2; ++nt) {
        _Float16* zr = &zbuf[wave][nt * 16 + n][0];
#pragma unroll
        for (int a = 0; a < 4; ++a) {
            const int m0 = 16 * a + 8 * g;
            v4f b0 = *(const v4f*)(bias_lds + 64 + m0);
            v4f b1 = *(const v4f*)(bias_lds + 64 + m0 + 4);
            v8h cz, sz;
#pragma unroll
            for (int r = 0; r < 8; ++r) {
                const float bv = (r < 4) ? b0[r] : b1[r - 4];
                const float yv = acc_y[nt][a][r] + bv;
                cz[r] = (_Float16)__cosf(yv);
                sz[r] = (_Float16)__sinf(yv);
            }
            *(v8h*)(zr + m0)      = cz;
            *(v8h*)(zr + 64 + m0) = sz;
        }
    }

    // ---- GEMM: k_out @ z2 ----
#pragma unroll
    for (int nt = 0; nt < 2; ++nt)
#pragma unroll
        for (int c = 0; c < 4; ++c)
            bz[nt][c] = *(const v16h*)(&zbuf[wave][nt * 16 + n][32 * c + 16 * g]);

#pragma unroll
    for (int nt = 0; nt < 2; ++nt)
#pragma unroll
        for (int a = 0; a < 4; ++a) acc_y[nt][a] = vzero;
#pragma unroll
    for (int c = 0; c < 4; ++c) {
#pragma unroll
        for (int a = 0; a < 4; ++a) {
            v16h ao = ldA(FRAG_OUT, a * 4 + c);
#pragma unroll
            for (int nt = 0; nt < 2; ++nt)
                acc_y[nt][a] = __builtin_amdgcn_wmma_f32_16x16x32_f16(
                    false, ao, false, bz[nt][c], (short)0, acc_y[nt][a], false, false);
        }
    }

    // ---- out = y_out + x_short + (b_out + b_short), coalesced f32 stores ----
#pragma unroll
    for (int nt = 0; nt < 2; ++nt) {
        const int pix = pixbase + nt * 16 + n;
#pragma unroll
        for (int a = 0; a < 4; ++a) {
            const int m0 = 16 * a + 8 * g;
            v4f bo0 = *(const v4f*)(bias_lds + 128 + m0);
            v4f bo1 = *(const v4f*)(bias_lds + 128 + m0 + 4);
            v4f bs0 = *(const v4f*)(bias_lds + 192 + m0);
            v4f bs1 = *(const v4f*)(bias_lds + 192 + m0 + 4);
#pragma unroll
            for (int r = 0; r < 8; ++r) {
                const float bsum = (r < 4) ? (bo0[r] + bs0[r]) : (bo1[r - 4] + bs1[r - 4]);
                const int m = m0 + r;
                ob[(size_t)m * HWPIX + pix] = acc_y[nt][a][r] + acc_s[nt][a][r] + bsum;
            }
        }
    }
}

// ---------------------------------------------------------------------------
extern "C" void kernel_launch(void* const* d_in, const int* in_sizes, int n_in,
                              void* d_out, int out_size, void* d_ws, size_t ws_size,
                              hipStream_t stream) {
    const float* x   = (const float*)d_in[0];   // [16][64][192][192]
    const float* lat = (const float*)d_in[1];   // [16][512]
    const float* Wk  = (const float*)d_in[2];   // [24832][512]
    const float* bk  = (const float*)d_in[3];   // [24832]
    float*       out = (float*)d_out;           // [16][64][192][192]

    _Float16* wfrag = (_Float16*)d_ws;                               // 786432 B
    float*    bias  = (float*)((char*)d_ws + BIAS_OFF_BYTES);        // 16 KiB

    dyn_weights_kernel<<<97, 256, 0, stream>>>(lat, Wk, bk, wfrag, bias);
    dyn_apply_kernel<<<dim3(16, 288), 128, 0, stream>>>(x, wfrag, bias, out);
}
